// FEAT_22488448762596
// MI455X (gfx1250) — matrix operand order repaired and checked by reading, not compiled, not run on hardware
//
#include <hip/hip_runtime.h>
#include <math.h>

typedef __bf16 bhalf;
typedef __attribute__((ext_vector_type(16))) __bf16 v16bf;
typedef __attribute__((ext_vector_type(8)))  __bf16 v8bf;   // 16 bytes: one ds_load_b128
typedef __attribute__((ext_vector_type(8)))  float  v8f;

// ---- model constants ----
#define BATCH  256
#define SEQ    2048
#define PATCHL 32
#define STRIDE 16
#define FN     127
#define NP     128
#define DMODEL 64
#define NH     4
#define DQK    16
#define DFF    256
#define DF2    128
#define NLAY   8
#define NCLS   11
#define INV_SCALE 0.25f   // 1/sqrt(DQK)

// ---- LDS strides (elements). All row strides are multiples of 8 elems (bf16)
// or 4 elems (f32) so every fragment run is 16-byte aligned -> ds_load_b128.
#define HS_LD  68   // f32 h master, 128 x 68         (68*4  = 272  B, 16B-mult)
#define HB_LD  72   // bf16 row-major bufs, 128 x 72  (72*2  = 144  B, 16B-mult)
#define WT_LD  136  // bf16 transposed weights, 64 x 136 (136*2 = 272 B)
#define VT_LD  136  // bf16 transposed V, 64 x 136
#define LG_LD  132  // f32 logits, 128 x 132          (132*4 = 528  B)
#define FB_LD  136  // bf16 FFN gate out (aliases LG), 128 x 136

#define OFF_HS 0
#define OFF_HB (OFF_HS + 128*HS_LD*4)        // 34816
#define OFF_QB (OFF_HB + 128*HB_LD*2)        // 53248
#define OFF_KB (OFF_QB + 128*HB_LD*2)        // 71680
#define OFF_VT (OFF_KB + 128*HB_LD*2)        // 90112  (64 x VT_LD bf16 = 17408 B)
#define OFF_LG (OFF_VT + 128*HB_LD*2)        // 108544
#define OFF_WT (OFF_LG + 128*LG_LD*4)        // 176128 (64 x WT_LD bf16 = 17408 B)
#define OFF_SM (OFF_WT + 128*HB_LD*2)        // 194560
#define SMEM_BYTES (OFF_SM + 64*4)           // 194816

// ====================== WMMA fragment loaders (wave32) ======================
// A-matrix 16x32 bf16 (ISA 7.12.2): m=lane&15; kb = (lane>=16)?8:0;
// elems 0..7 -> K=kb..kb+7, elems 8..15 -> K=kb+16..kb+23. Two 16B runs.
__device__ __forceinline__ v16bf ldsA_ct(const bhalf* base, int ld, int lane) {
    const int m  = lane & 15;
    const int kb = (lane & 16) ? 8 : 0;
    const bhalf* p = base + m * ld + kb;
    const v8bf lo = *(const v8bf*)(p);        // K = kb .. kb+7
    const v8bf hi = *(const v8bf*)(p + 16);   // K = kb+16 .. kb+23
    v16bf a;
#pragma unroll
    for (int j = 0; j < 8; ++j) { a[j] = lo[j]; a[j + 8] = hi[j]; }
    return a;
}

// A fragment from f32 LDS (attention probs), cvt to bf16; two 32B runs.
__device__ __forceinline__ v16bf ldsA_f32(const float* base, int ld, int lane) {
    const int m  = lane & 15;
    const int kb = (lane & 16) ? 8 : 0;
    const float* p = base + m * ld + kb;
    const float4 a0 = *(const float4*)(p);
    const float4 a1 = *(const float4*)(p + 4);
    const float4 b0 = *(const float4*)(p + 16);
    const float4 b1 = *(const float4*)(p + 20);
    v16bf a;
    a[0]=(bhalf)a0.x; a[1]=(bhalf)a0.y; a[2]=(bhalf)a0.z; a[3]=(bhalf)a0.w;
    a[4]=(bhalf)a1.x; a[5]=(bhalf)a1.y; a[6]=(bhalf)a1.z; a[7]=(bhalf)a1.w;
    a[8]=(bhalf)b0.x; a[9]=(bhalf)b0.y; a[10]=(bhalf)b0.z; a[11]=(bhalf)b0.w;
    a[12]=(bhalf)b1.x; a[13]=(bhalf)b1.y; a[14]=(bhalf)b1.z; a[15]=(bhalf)b1.w;
    return a;
}

// A fragment from bf16 Q with per-head-block wmix scale folded in.
// Each contiguous 8-elem run lies inside a single 16-wide head block.
__device__ __forceinline__ v16bf ldsA_qmix(const bhalf* base, int ld, int lane,
                                           const float* wmixS, int g, int k0) {
    const int m  = lane & 15;
    const int kb = (lane & 16) ? 8 : 0;
    const bhalf* p = base + m * ld + kb;
    const v8bf lo = *(const v8bf*)(p);
    const v8bf hi = *(const v8bf*)(p + 16);
    const float s0 = wmixS[((k0 + kb) >> 4) * NH + g];
    const float s1 = wmixS[((k0 + kb + 16) >> 4) * NH + g];
    v16bf a;
#pragma unroll
    for (int j = 0; j < 8; ++j) {
        a[j]     = (bhalf)((float)lo[j] * s0);
        a[j + 8] = (bhalf)((float)hi[j] * s1);
    }
    return a;
}

// B-matrix 32x16 bf16 from a TRANSPOSED source: logical B[k][n] = src[n*ld + k].
// lane n=lane&15; kb=(lane>=16)?8:0; elems 0..7 -> K=kb..kb+7, elems 8..15 -> K=kb+16..kb+23.
__device__ __forceinline__ v16bf ldsB_ct(const bhalf* base, int ld, int lane) {
    const int n  = lane & 15;
    const int k0 = (lane & 16) ? 8 : 0;
    const bhalf* p = base + n * ld + k0;
    const v8bf lo = *(const v8bf*)(p);
    const v8bf hi = *(const v8bf*)(p + 16);
    v16bf b;
#pragma unroll
    for (int j = 0; j < 8; ++j) { b[j] = lo[j]; b[j + 8] = hi[j]; }
    return b;
}

#define WMMA_BF16(a, b, c) \
    __builtin_amdgcn_wmma_f32_16x16x32_bf16(false, (a), false, (b), (short)0, (c), false, false)

__device__ __forceinline__ float gelu_exact(float v) {
    return 0.5f * v * (1.0f + erff(v * 0.70710678118654752f));
}

__device__ __forceinline__ void ln_row(float* row, const float* g, const float* b, bhalf* brow) {
    float mu = 0.f;
    for (int d = 0; d < DMODEL; ++d) mu += row[d];
    mu *= (1.0f / DMODEL);
    float var = 0.f;
    for (int d = 0; d < DMODEL; ++d) { float t = row[d] - mu; var += t * t; }
    var *= (1.0f / DMODEL);
    const float inv = rsqrtf(var + 1e-5f);
    for (int d = 0; d < DMODEL; ++d) {
        float v = (row[d] - mu) * inv * g[d] + b[d];
        row[d] = v;
        brow[d] = (bhalf)v;
    }
}

// =============================== fused kernel ===============================
__global__ __launch_bounds__(256)
void vit_fused(const float* __restrict__ x,       const float* __restrict__ frame_W,
               const float* __restrict__ cls_tok, const float* __restrict__ pos_emb,
               const float* __restrict__ Wq,      const float* __restrict__ Wk,
               const float* __restrict__ Wv,      const float* __restrict__ Wmix,
               const float* __restrict__ Wo,      const float* __restrict__ ln1g,
               const float* __restrict__ ln1b,    const float* __restrict__ ln2g,
               const float* __restrict__ ln2b,    const float* __restrict__ W1,
               const float* __restrict__ b1,      const float* __restrict__ W2,
               const float* __restrict__ b2,      const float* __restrict__ Wc,
               const float* __restrict__ bc,
               float* __restrict__ out, float* __restrict__ arws)
{
    extern __shared__ char smem[];
    float* hS = (float*)(smem + OFF_HS);   // 128 x HS_LD f32 : master activations
    bhalf* hB = (bhalf*)(smem + OFF_HB);   // 128 x HB_LD bf16: h (WMMA A) / attn-out O
    bhalf* qB = (bhalf*)(smem + OFF_QB);   // 128 x HB_LD bf16: Q (pre-scaled), row-major
    bhalf* kB = (bhalf*)(smem + OFF_KB);   // 128 x HB_LD bf16: K, row-major (read as B^T)
    bhalf* vT = (bhalf*)(smem + OFF_VT);   // 64  x VT_LD bf16: V TRANSPOSED [d][j]
    float* lg = (float*)(smem + OFF_LG);   // 128 x LG_LD f32 : logits / probs
    bhalf* fB = (bhalf*)(smem + OFF_LG);   // aliases lg: 128 x FB_LD bf16 FFN gate out
    bhalf* wT = (bhalf*)(smem + OFF_WT);   // 64  x WT_LD bf16: weights TRANSPOSED [n][k]
    float* sm = (float*)(smem + OFF_SM);   // small scratch: wmix[16] + classifier

    const int bidx = blockIdx.x;
    const int tid  = threadIdx.x;
    const int lane = tid & 31;
    const int wv   = tid >> 5;         // wave id 0..7
    const int m0   = wv * 16;          // this wave's M-strip
    const int mHi  = (lane & 16) ? 8 : 0;
    const int nLo  = lane & 15;

    float* ar = arws + (size_t)bidx * (NH * NP * NP);   // attention residual (global)

    // ---------------- patch embedding: h = patches @ frame_W + pos (+cls) ----------------
    // frame_W is (K=64 x N=64): stage transposed wT[n][k]
    for (int i = tid; i < 64 * 64; i += 256)
        wT[(i & 63) * WT_LD + (i >> 6)] = (bhalf)frame_W[i];
    for (int i = tid; i < 128 * 64; i += 256) {
        const int r = i >> 6, c = i & 63;
        float v = 0.f;
        if (r < FN) {
            const int ch = c >> 5, p = c & 31;
            v = x[((size_t)bidx * SEQ + r * STRIDE + p) * 2 + ch];
        }
        qB[r * HB_LD + c] = (bhalf)v;   // stage patches in qB (free pre-attention)
    }
    __syncthreads();

    for (int nt = 0; nt < 4; ++nt) {
        v8f acc = {};
#pragma unroll
        for (int kt = 0; kt < 2; ++kt) {
            v16bf a = ldsA_ct(qB + m0 * HB_LD + kt * 32, HB_LD, lane);
            v16bf b = ldsB_ct(wT + (nt * 16) * WT_LD + kt * 32, WT_LD, lane);
            acc = WMMA_BF16(a, b, acc);
        }
#pragma unroll
        for (int r = 0; r < 8; ++r) {
            const int m = m0 + r + mHi;
            const int n = nt * 16 + nLo;
            float hv = acc[r] + pos_emb[m * DMODEL + n] + (m == NP - 1 ? cls_tok[n] : 0.f);
            hS[m * HS_LD + n] = hv;
            hB[m * HB_LD + n] = (bhalf)hv;
        }
    }
    __syncthreads();

    // ================================ layer loop ================================
    for (int l = 0; l < NLAY; ++l) {
        const float* wq = Wq + l * 64 * 64;
        const float* wk = Wk + l * 64 * 64;
        const float* wvv = Wv + l * 64 * 64;
        const float* wo = Wo + l * 64 * 64;
        const float* w1 = W1 + l * 64 * 256;
        const float* w2 = W2 + l * 128 * 64;

        if (lane == 0) {                       // warm GL2 for upcoming weights
            __builtin_prefetch(w1, 0, 1);
            __builtin_prefetch(w2, 0, 1);
        }

        // ---- Q/K/V projections (A = hB, B = staged transposed weight) ----
        const float* proj[3] = {wq, wk, wvv};
        for (int pidx = 0; pidx < 3; ++pidx) {
            for (int i = tid; i < 64 * 64; i += 256)
                wT[(i & 63) * WT_LD + (i >> 6)] = (bhalf)proj[pidx][i];
            if (pidx == 0 && tid < 16) sm[tid] = Wmix[l * 16 + tid];
            __syncthreads();
            const float oscale = (pidx == 0) ? INV_SCALE : 1.0f;
            for (int nt = 0; nt < 4; ++nt) {
                v8f acc = {};
#pragma unroll
                for (int kt = 0; kt < 2; ++kt) {
                    v16bf a = ldsA_ct(hB + m0 * HB_LD + kt * 32, HB_LD, lane);
                    v16bf b = ldsB_ct(wT + (nt * 16) * WT_LD + kt * 32, WT_LD, lane);
                    acc = WMMA_BF16(a, b, acc);
                }
#pragma unroll
                for (int r = 0; r < 8; ++r) {
                    const int m = m0 + r + mHi;
                    const int n = nt * 16 + nLo;
                    const bhalf val = (bhalf)(acc[r] * oscale);
                    if (pidx == 0)      qB[m * HB_LD + n] = val;
                    else if (pidx == 1) kB[m * HB_LD + n] = val;
                    else                vT[n * VT_LD + m] = val;   // V transposed [d][j]
                }
            }
            __syncthreads();
        }

        // ---- per output-head: mixed logits -> (+residual) softmax -> O_g ----
        for (int g = 0; g < NH; ++g) {
            v16bf afr[2];
#pragma unroll
            for (int kt = 0; kt < 2; ++kt)
                afr[kt] = ldsA_qmix(qB + m0 * HB_LD + kt * 32, HB_LD, lane, sm, g, kt * 32);
            for (int nt = 0; nt < 8; ++nt) {
                v8f acc = {};
#pragma unroll
                for (int kt = 0; kt < 2; ++kt) {
                    // B[k][n] = kB[(nt*16+n)*HB_LD + k] : contiguous in k
                    v16bf b = ldsB_ct(kB + (nt * 16) * HB_LD + kt * 32, HB_LD, lane);
                    acc = WMMA_BF16(afr[kt], b, acc);
                }
#pragma unroll
                for (int r = 0; r < 8; ++r) {
                    const int i = m0 + r + mHi;
                    const int j = nt * 16 + nLo;
                    float v = acc[r];
                    if (l > 0) v += ar[(g * NP + i) * NP + j];   // attention residual
                    lg[i * LG_LD + j] = v;
                }
            }
            __syncthreads();

            // row-wise softmax in LDS
            if (tid < NP) {
                float* row = lg + tid * LG_LD;
                float mx = -1e30f;
                for (int j = 0; j < NP; ++j) mx = fmaxf(mx, row[j]);
                float s = 0.f;
                for (int j = 0; j < NP; ++j) { float e = __expf(row[j] - mx); row[j] = e; s += e; }
                const float inv = 1.0f / s;
                for (int j = 0; j < NP; ++j) row[j] *= inv;
            }
            __syncthreads();

            // persist probs as next layer's residual
            for (int i = tid; i < NP * NP; i += 256)
                ar[g * (NP * NP) + i] = lg[(i >> 7) * LG_LD + (i & 127)];

            // O_g = probs @ V_g  -> hB columns [g*16, g*16+16)
            {
                v8f acc = {};
#pragma unroll
                for (int kt = 0; kt < 4; ++kt) {
                    v16bf a = ldsA_f32(lg + m0 * LG_LD + kt * 32, LG_LD, lane);
                    // B[k][n] = V[k][g*16+n] = vT[(g*16+n)*VT_LD + k] : contiguous in k
                    v16bf b = ldsB_ct(vT + (g * 16) * VT_LD + kt * 32, VT_LD, lane);
                    acc = WMMA_BF16(a, b, acc);
                }
#pragma unroll
                for (int r = 0; r < 8; ++r)
                    hB[(m0 + r + mHi) * HB_LD + g * 16 + nLo] = (bhalf)acc[r];
            }
            __syncthreads();
        }

        // ---- h = LN1(h + O @ Wo) ----
        for (int i = tid; i < 64 * 64; i += 256)
            wT[(i & 63) * WT_LD + (i >> 6)] = (bhalf)wo[i];
        __syncthreads();
        for (int nt = 0; nt < 4; ++nt) {
            v8f acc = {};
#pragma unroll
            for (int kt = 0; kt < 2; ++kt) {
                v16bf a = ldsA_ct(hB + m0 * HB_LD + kt * 32, HB_LD, lane);
                v16bf b = ldsB_ct(wT + (nt * 16) * WT_LD + kt * 32, WT_LD, lane);
                acc = WMMA_BF16(a, b, acc);
            }
#pragma unroll
            for (int r = 0; r < 8; ++r)
                hS[(m0 + r + mHi) * HS_LD + nt * 16 + nLo] += acc[r];
        }
        __syncthreads();
        if (tid < NP)
            ln_row(hS + tid * HS_LD, ln1g + l * DMODEL, ln1b + l * DMODEL, hB + tid * HB_LD);
        __syncthreads();

        // ---- gated FFN: u = h@W1+b1 ; f = u1*gelu(u2)+u2*gelu(u1) ----
        const float* bb1 = b1 + l * DFF;
        for (int jt = 0; jt < 8; ++jt) {
            // stage paired W1 col-blocks [jt*16) and [128+jt*16) transposed: 32 x 64
            for (int i = tid; i < 64 * 32; i += 256) {
                const int r = i >> 5, c = i & 31;
                const int col = (c < 16) ? (jt * 16 + c) : (DF2 + jt * 16 + (c - 16));
                wT[c * WT_LD + r] = (bhalf)w1[r * DFF + col];
            }
            __syncthreads();
            v8f a1 = {}, a2 = {};
#pragma unroll
            for (int kt = 0; kt < 2; ++kt) {
                v16bf a  = ldsA_ct(hB + m0 * HB_LD + kt * 32, HB_LD, lane);
                v16bf bl = ldsB_ct(wT + 0  * WT_LD + kt * 32, WT_LD, lane);
                v16bf br = ldsB_ct(wT + 16 * WT_LD + kt * 32, WT_LD, lane);
                a1 = WMMA_BF16(a, bl, a1);
                a2 = WMMA_BF16(a, br, a2);
            }
#pragma unroll
            for (int r = 0; r < 8; ++r) {
                const int m = m0 + r + mHi;
                const float u1 = a1[r] + bb1[jt * 16 + nLo];
                const float u2 = a2[r] + bb1[DF2 + jt * 16 + nLo];
                fB[m * FB_LD + jt * 16 + nLo] = (bhalf)(u1 * gelu_exact(u2) + u2 * gelu_exact(u1));
            }
            __syncthreads();
        }

        // ---- h = LN2(h + f @ W2 + b2) ----
        for (int i = tid; i < 128 * 64; i += 256)
            wT[(i & 63) * WT_LD + (i >> 6)] = (bhalf)w2[i];   // W2: 128x64 -> wT 64x128
        __syncthreads();
        for (int nt = 0; nt < 4; ++nt) {
            v8f acc = {};
#pragma unroll
            for (int kt = 0; kt < 4; ++kt) {
                v16bf a = ldsA_ct(fB + m0 * FB_LD + kt * 32, FB_LD, lane);
                v16bf b = ldsB_ct(wT + (nt * 16) * WT_LD + kt * 32, WT_LD, lane);
                acc = WMMA_BF16(a, b, acc);
            }
#pragma unroll
            for (int r = 0; r < 8; ++r) {
                const int m = m0 + r + mHi;
                const int n = nt * 16 + nLo;
                hS[m * HS_LD + n] += acc[r] + b2[l * DMODEL + n];
            }
        }
        __syncthreads();
        if (tid < NP)
            ln_row(hS + tid * HS_LD, ln2g + l * DMODEL, ln2b + l * DMODEL, hB + tid * HB_LD);
        __syncthreads();
    }

    // ---------------- classifier on CLS row (row 127) ----------------
    if (tid < NCLS) {
        float s = bc[tid];
        const float* hrow = hS + (NP - 1) * HS_LD;
        for (int d = 0; d < DMODEL; ++d) s += hrow[d] * Wc[d * NCLS + tid];
        sm[32 + tid] = s;
    }
    __syncthreads();
    if (tid == 0) {
        float mx = -1e30f;
        for (int c = 0; c < NCLS; ++c) mx = fmaxf(mx, sm[32 + c]);
        float e[NCLS], ssum = 0.f;
        for (int c = 0; c < NCLS; ++c) { e[c] = __expf(sm[32 + c] - mx); ssum += e[c]; }
        const float inv = 1.0f / ssum;
        for (int c = 0; c < NCLS; ++c) out[bidx * NCLS + c] = e[c] * inv;
    }
}

extern "C" void kernel_launch(void* const* d_in, const int* in_sizes, int n_in,
                              void* d_out, int out_size, void* d_ws, size_t ws_size,
                              hipStream_t stream) {
    const float* x        = (const float*)d_in[0];
    const float* frame_W  = (const float*)d_in[1];
    const float* cls_tok  = (const float*)d_in[2];
    const float* pos_emb  = (const float*)d_in[3];
    const float* Wq       = (const float*)d_in[4];
    const float* Wk       = (const float*)d_in[5];
    const float* Wv       = (const float*)d_in[6];
    const float* Wmix     = (const float*)d_in[7];
    const float* Wo       = (const float*)d_in[8];
    const float* ln1g     = (const float*)d_in[9];
    const float* ln1b     = (const float*)d_in[10];
    const float* ln2g     = (const float*)d_in[11];
    const float* ln2b     = (const float*)d_in[12];
    const float* W1       = (const float*)d_in[13];
    const float* b1       = (const float*)d_in[14];
    const float* W2       = (const float*)d_in[15];
    const float* b2       = (const float*)d_in[16];
    const float* Wc       = (const float*)d_in[17];
    const float* bc       = (const float*)d_in[18];
    float* out = (float*)d_out;
    float* ar  = (float*)d_ws;   // attention residual: B*H*NP*NP f32 = 64 MB

    (void)in_sizes; (void)n_in; (void)out_size; (void)ws_size;

    hipFuncSetAttribute((const void*)vit_fused,
                        hipFuncAttributeMaxDynamicSharedMemorySize, SMEM_BYTES);

    vit_fused<<<BATCH, 256, SMEM_BYTES, stream>>>(
        x, frame_W, cls_tok, pos_emb, Wq, Wk, Wv, Wmix, Wo,
        ln1g, ln1b, ln2g, ln2b, W1, b1, W2, b2, Wc, bc, out, ar);
}
